// DelayNetwork_62036507623824
// MI455X (gfx1250) — matrix operand order repaired and checked by reading, not compiled, not run on hardware
//
#include <hip/hip_runtime.h>
#include <hip/hip_bf16.h>
#include <math.h>

// ---------------------------------------------------------------------------
// DelayNetwork SNN forward for MI455X (gfx1250, wave32, WMMA f16->f32)
//
// Sizes (compile-time constants matching the reference):
//   N_IN=700 (padded 704), N_REC=2048, N_OUT=20, BATCH=32, T=500, D_MAX=8
//
// Strategy:
//  * hoist inp_cur = x@W_i + B_r out of the scan -> one parallel WMMA GEMM
//  * per step: rec_cur = concat_k(spk[t-k]) @ concat_k(W_r*mask_k)
//    as a single M=32, K=16384, N=2048 WMMA GEMM (K split by delay over
//    8 waves/block, LDS reduce). Masked weight stack pre-packed in f16
//    WMMA-B layout (67 MB -> L2 resident across all 500 steps).
//  * spike history kept as f16 ring buffer (16 slots) in WMMA-friendly
//    (slot, b, r) layout; fp32 spikes written straight to d_out raster.
// ---------------------------------------------------------------------------

typedef __attribute__((ext_vector_type(16))) _Float16 v16h;
typedef __attribute__((ext_vector_type(8)))  _Float16 v8h;
typedef __attribute__((ext_vector_type(8)))  float    v8f;

#define N_IN    700
#define N_INP   704          // padded to multiple of 32
#define KC_IN   22           // 704/32
#define N_REC   2048
#define N_OUT   20
#define BATCH   32
#define T_STEPS 500
#define D_MAX   8
#define K_REC   (D_MAX * N_REC)   // 16384
#define KC_REC  (K_REC / 32)      // 512
#define NT      (N_REC / 16)      // 128 column tiles
#define RING    16                // ring slots (power of two >= D_MAX+1)

// d_out layout (fp32, concatenated return order):
//   mem_out        : 32*20                     @ 0
//   rec_spike_rast : 32*2048*500 (b,r,t)       @ 640
//   mem_out_hist   : 32*500*20   (b,t,o)       @ 640 + 32768000
#define OUT_RASTER 640
#define OUT_HIST   (640 + BATCH * N_REC * T_STEPS)

// workspace byte offsets (all 256-aligned)
#define OFF_XT    0ull                       // f16 [T][32][704]      22,528,000
#define OFF_WIP   22528000ull                // f16 [128][22][32][16]  2,883,584
#define OFF_INP   25411584ull                // f16 [T][32][2048]     65,536,000
#define OFF_WSP   90947584ull                // f16 [128][512][32][16] 67,108,864
#define OFF_RING  158056448ull               // f16 [16][32][2048]     2,097,152
#define OFF_MREC  160153600ull               // f32 [32][2048]           262,144
#define OFF_MOUT  160415744ull               // f32 [32][20]               2,560

__device__ __forceinline__ v16h cat8(v8h lo, v8h hi) {
  return __builtin_shufflevector(lo, hi, 0,1,2,3,4,5,6,7,8,9,10,11,12,13,14,15);
}

// --------------------------- prep kernels ----------------------------------

// zero mem_rec + mem_out state
__global__ void k_init(float* mrec, float* mout) {
  int i = blockIdx.x * blockDim.x + threadIdx.x;
  if (i < BATCH * N_REC) mrec[i] = 0.0f;
  if (i < BATCH * N_OUT) mout[i] = 0.0f;
}

// raster (b, i, t) f32  ->  xT16 (t, b, ipad) f16, zero pad i in [700,704)
__global__ void k_pack_x(const float* __restrict__ raster, _Float16* __restrict__ xT) {
  int idx = blockIdx.x * blockDim.x + threadIdx.x;
  if (idx >= T_STEPS * BATCH * N_INP) return;
  int ip = idx % N_INP;
  int b  = (idx / N_INP) % BATCH;
  int t  = idx / (N_INP * BATCH);
  float v = (ip < N_IN) ? raster[(size_t)b * N_IN * T_STEPS + (size_t)ip * T_STEPS + t] : 0.0f;
  xT[idx] = (_Float16)v;
}

// W_i (K=700, N=2048) f32 -> WMMA-B-layout f16 [ntile][kc][lane][16]
__global__ void k_pack_wi(const float* __restrict__ Wi, _Float16* __restrict__ WiP) {
  int idx = blockIdx.x * blockDim.x + threadIdx.x;
  if (idx >= NT * KC_IN * 32 * 16) return;
  int h     = idx & 15;
  int lane  = (idx >> 4) & 31;
  int kc    = (idx >> 9) % KC_IN;
  int ntile = idx / (16 * 32 * KC_IN);
  int n = ntile * 16 + (lane & 15);
  int k = kc * 32 + ((lane & 16) ? 8 : 0) + (h < 8 ? h : h + 8);
  float v = (k < N_IN) ? Wi[(size_t)k * N_REC + n] : 0.0f;
  WiP[idx] = (_Float16)v;
}

// masked W_r stack -> WMMA-B-layout f16 [ntile][kc(512)][lane][16]
// global K = kd*2048 + r, value = mask[kd][r][n] ? W_r[r][n] : 0
__global__ void k_pack_ws(const float* __restrict__ Wr,
                          const unsigned char* __restrict__ mask,  // bool (K,R,R)
                          _Float16* __restrict__ WsP) {
  int idx = blockIdx.x * blockDim.x + threadIdx.x;   // 33,554,432 threads
  int h     = idx & 15;
  int lane  = (idx >> 4) & 31;
  int kc    = (idx >> 9) & 511;
  int ntile = idx >> 18;
  int n  = ntile * 16 + (lane & 15);
  int k  = kc * 32 + ((lane & 16) ? 8 : 0) + (h < 8 ? h : h + 8);
  int kd = k >> 11;          // delay class index 0..7 (delay = kd+1)
  int r  = k & 2047;
  float w = Wr[(size_t)r * N_REC + n];
  unsigned char m = mask[(size_t)kd * N_REC * N_REC + (size_t)r * N_REC + n];
  WsP[idx] = (_Float16)(m ? w : 0.0f);
}

// ----------------------- input-current GEMM (parallel) ---------------------
// inp_cur[t] = x_t @ W_i + B_r   (M=32,K=704,N=2048 per t; 500 t's)
__global__ void k_inp_gemm(const _Float16* __restrict__ xT,
                           const _Float16* __restrict__ WiP,
                           const float* __restrict__ Br,
                           _Float16* __restrict__ inp16) {
  int wave = blockIdx.x * (blockDim.x >> 5) + (threadIdx.x >> 5);
  int lane = threadIdx.x & 31;
  int t     = wave >> 8;            // 256 tiles per t
  int q     = wave & 255;
  int mtile = q & 1;
  int ntile = q >> 1;
  int b     = mtile * 16 + (lane & 15);
  int kbase = (lane & 16) ? 8 : 0;

  const _Float16* arow = xT + ((size_t)t * BATCH + b) * N_INP;
  v8f c = {};
  for (int kc = 0; kc < KC_IN; ++kc) {
    v8h lo = *(const v8h*)(arow + kc * 32 + kbase);
    v8h hi = *(const v8h*)(arow + kc * 32 + 16 + kbase);
    v16h a = cat8(lo, hi);
    v16h bm = *(const v16h*)(WiP + (((size_t)ntile * KC_IN + kc) * 32 + lane) * 16);
    c = __builtin_amdgcn_wmma_f32_16x16x32_f16(false, a, false, bm, (short)0, c, false, false);
  }
  int n = ntile * 16 + (lane & 15);
  float bias = Br[n];
#pragma unroll
  for (int v = 0; v < 8; ++v) {
    int bb = mtile * 16 + v + ((lane & 16) ? 8 : 0);
    inp16[((size_t)t * BATCH + bb) * N_REC + n] = (_Float16)(c[v] + bias);
  }
}

// ----------------------- per-step recurrent WMMA GEMM ----------------------
// one block = one 16x16 output tile; 8 waves, wave w handles delay d=w+1
// (64 K-chunks each), then LDS reduce + pointwise LIF update.
__global__ void k_step(int t,
                       const _Float16* __restrict__ WsP,
                       const _Float16* __restrict__ inp16,
                       _Float16* __restrict__ ring,
                       float* __restrict__ mrec,
                       const float* __restrict__ tau_rec,
                       const float* __restrict__ thr_rec,
                       float* __restrict__ out) {
  __shared__ float partial[D_MAX][256];
  int tile  = blockIdx.x;
  int mtile = tile & 1;
  int ntile = tile >> 1;
  int w     = threadIdx.x >> 5;     // delay index 0..7, delay = w+1
  int lane  = threadIdx.x & 31;
  int b     = mtile * 16 + (lane & 15);
  int kbase = (lane & 16) ? 8 : 0;

  v8f c = {};
  int ts = t - 1 - w;               // source spike time; valid iff ts >= 1
  if (ts >= 1) {                    // wave-uniform branch: EXEC stays full
    const _Float16* arow = ring + (((size_t)(ts & (RING - 1)) * BATCH + b) * N_REC);
    const _Float16* bbase = WsP + (((size_t)ntile * KC_REC + (size_t)w * 64) * 32 + lane) * 16;
    for (int j = 0; j < 64; ++j) {
      v8h lo = *(const v8h*)(arow + j * 32 + kbase);
      v8h hi = *(const v8h*)(arow + j * 32 + 16 + kbase);
      v16h a = cat8(lo, hi);
      v16h bm = *(const v16h*)(bbase + (size_t)j * 32 * 16);
      c = __builtin_amdgcn_wmma_f32_16x16x32_f16(false, a, false, bm, (short)0, c, false, false);
    }
  }
#pragma unroll
  for (int v = 0; v < 8; ++v) partial[w][v * 32 + lane] = c[v];
  __syncthreads();

  // 256 threads -> 256 tile elements
  int tid   = threadIdx.x;
  float sum = 0.0f;
#pragma unroll
  for (int k = 0; k < D_MAX; ++k) sum += partial[k][tid];
  int v     = tid >> 5;
  int laneE = tid & 31;
  int M     = v + ((laneE & 16) ? 8 : 0);
  int N     = laneE & 15;
  int bb    = mtile * 16 + M;
  int nn    = ntile * 16 + N;

  float rc  = __expf(-0.01f / tau_rec[0]);
  float thr = thr_rec[0];
  size_t si = (size_t)bb * N_REC + nn;
  float mem = mrec[si] * rc + (float)inp16[((size_t)t * BATCH + bb) * N_REC + nn] + sum;
  float spk = (mem - thr > 0.0f) ? 1.0f : 0.0f;
  mem -= spk * thr;
  mrec[si] = mem;
  ring[(((size_t)(t & (RING - 1)) * BATCH + bb) * N_REC) + nn] = (_Float16)spk;
  out[OUT_RASTER + ((size_t)bb * N_REC + nn) * T_STEPS + t] = spk;
}

// ----------------------- per-step output projection ------------------------
__global__ void k_step_out(int t,
                           const _Float16* __restrict__ ring,
                           const float* __restrict__ Wo,
                           const float* __restrict__ Bo,
                           const float* __restrict__ tau_out,
                           float* __restrict__ mout,
                           float* __restrict__ out) {
  int idx = blockIdx.x * blockDim.x + threadIdx.x;
  if (idx >= BATCH * N_OUT) return;
  int b = idx / N_OUT;
  int o = idx % N_OUT;
  const _Float16* s = ring + ((size_t)(t & (RING - 1)) * BATCH + b) * N_REC;
  float dot = 0.0f;
  for (int r = 0; r < N_REC; ++r) dot += (float)s[r] * Wo[(size_t)r * N_OUT + o];
  float oc = __expf(-0.01f / tau_out[0]);
  float m  = mout[idx] * oc + dot + Bo[o];
  mout[idx] = m;
  out[OUT_HIST + ((size_t)b * T_STEPS + t) * N_OUT + o] = m;
  if (t == T_STEPS - 1) out[idx] = m;   // final mem_out
}

// ---------------------------------------------------------------------------
extern "C" void kernel_launch(void* const* d_in, const int* in_sizes, int n_in,
                              void* d_out, int out_size, void* d_ws, size_t ws_size,
                              hipStream_t stream) {
  const float*         raster  = (const float*)d_in[0];
  // d_in[1] = delay_values (1..8) — implied by mask stacking order, unused
  const unsigned char* masks   = (const unsigned char*)d_in[2];  // bool (8,R,R)
  const float*         Wi      = (const float*)d_in[3];
  const float*         Wr      = (const float*)d_in[4];
  const float*         Wo      = (const float*)d_in[5];
  const float*         Br      = (const float*)d_in[6];
  const float*         Bo      = (const float*)d_in[7];
  const float*         tau_rec = (const float*)d_in[8];
  const float*         tau_out = (const float*)d_in[9];
  const float*         thr_rec = (const float*)d_in[10];
  float*               out     = (float*)d_out;

  char* ws = (char*)d_ws;
  _Float16* xT    = (_Float16*)(ws + OFF_XT);
  _Float16* WiP   = (_Float16*)(ws + OFF_WIP);
  _Float16* inp16 = (_Float16*)(ws + OFF_INP);
  _Float16* WsP   = (_Float16*)(ws + OFF_WSP);
  _Float16* ring  = (_Float16*)(ws + OFF_RING);
  float*    mrec  = (float*)(ws + OFF_MREC);
  float*    mout  = (float*)(ws + OFF_MOUT);

  // state init (idempotent per launch)
  k_init<<<(BATCH * N_REC + 255) / 256, 256, 0, stream>>>(mrec, mout);

  // pack operands into WMMA layouts
  k_pack_x <<<(T_STEPS * BATCH * N_INP + 255) / 256, 256, 0, stream>>>(raster, xT);
  k_pack_wi<<<(NT * KC_IN * 32 * 16 + 255) / 256, 256, 0, stream>>>(Wi, WiP);
  k_pack_ws<<<(NT * KC_REC * 32 * 16) / 256, 256, 0, stream>>>(Wr, masks, WsP);

  // hoisted input-current GEMM: 500 * 256 tiles, 8 waves/block
  k_inp_gemm<<<(T_STEPS * 256) / 8, 256, 0, stream>>>(xT, WiP, Br, inp16);

  // sequential scan: 500 steps, each = recurrent WMMA GEMM + output proj
  for (int t = 0; t < T_STEPS; ++t) {
    k_step<<<256, 256, 0, stream>>>(t, WsP, inp16, ring, mrec, tau_rec, thr_rec, out);
    k_step_out<<<(BATCH * N_OUT + 63) / 64, 64, 0, stream>>>(t, ring, Wo, Bo, tau_out, mout, out);
  }
}
